// EncoderRNN_51230369907534
// MI455X (gfx1250) — matrix operand order repaired and checked by reading, not compiled, not run on hardware
//
#include <hip/hip_runtime.h>

typedef __bf16 bf16;
typedef __attribute__((ext_vector_type(16))) bf16  v16bf;
typedef __attribute__((ext_vector_type(8)))  bf16  v8bf;
typedef __attribute__((ext_vector_type(4)))  bf16  v4bf;
typedef __attribute__((ext_vector_type(8)))  float v8f;

static constexpr int BATCH = 64;
static constexpr int TLEN  = 512;
static constexpr int EMBD  = 512;
static constexpr int HIDD  = 1024;
static constexpr int G4    = 4096;   // 4*HIDD

// ---------------------------------------------------------------------------
// Prep: f32 -> bf16 weight conversion, bias fusion, state zero-init.
// ---------------------------------------------------------------------------
__global__ void prep_kernel(const float* __restrict__ wih_f,
                            const float* __restrict__ whh_f,
                            const float* __restrict__ bih,
                            const float* __restrict__ bhh,
                            bf16* __restrict__ wih_b,
                            bf16* __restrict__ whh_b,
                            float* __restrict__ bsum,
                            bf16* __restrict__ h0_b,
                            float* __restrict__ cst)
{
    size_t i      = (size_t)blockIdx.x * blockDim.x + threadIdx.x;
    size_t stride = (size_t)gridDim.x * blockDim.x;
    for (size_t k = i; k < (size_t)G4 * HIDD; k += stride) whh_b[k] = (bf16)whh_f[k];
    for (size_t k = i; k < (size_t)G4 * EMBD; k += stride) wih_b[k] = (bf16)wih_f[k];
    for (size_t k = i; k < (size_t)G4;        k += stride) bsum[k]  = bih[k] + bhh[k];
    for (size_t k = i; k < (size_t)BATCH * HIDD; k += stride) {
        h0_b[k] = (bf16)0.0f;
        cst[k]  = 0.0f;
    }
}

// ---------------------------------------------------------------------------
// Embedding gather + bf16 convert.  Row layout: xemb[(t*64 + b)*512 + e]
// ---------------------------------------------------------------------------
__global__ __launch_bounds__(128)
void embed_kernel(const int* __restrict__ sent,      // [64, 512]
                  const float* __restrict__ table,   // [32000, 512]
                  bf16* __restrict__ xemb)           // [T*64, 512]
{
    const int row = blockIdx.x;          // t*64 + b
    const int t   = row >> 6;
    const int b   = row & 63;
    const int tok = sent[b * TLEN + t];
    const float4* __restrict__ src = (const float4*)(table + (size_t)tok * EMBD);
    v4bf* __restrict__ dst = (v4bf*)(xemb + (size_t)row * EMBD);
    for (int i = threadIdx.x; i < EMBD / 4; i += 128) {
        float4 f = src[i];
        v4bf o;
        o[0] = (bf16)f.x; o[1] = (bf16)f.y; o[2] = (bf16)f.z; o[3] = (bf16)f.w;
        dst[i] = o;
    }
}

// ---------------------------------------------------------------------------
// Fragment loaders (CDNA5 bf16 WMMA VGPR layout, wave32).
// A (16x32 MxK): lane L -> m = L%16, half = L/16
//   frag elems 0..7  = A[m][kc + half*8      + i]
//   frag elems 8..15 = A[m][kc + 16 + half*8 + i]
// B (32x16 KxN): lane L -> n = L%16, half = L/16
//   frag elems 0..7 = Bt_row_n[kc + half*8 + i], 8..15 = Bt_row_n[kc + 16 + half*8 + i]   (row of W)
// ---------------------------------------------------------------------------
__device__ __forceinline__ v16bf load_a_frag(const bf16* __restrict__ row, int kc, int half)
{
    v8bf lo = *(const v8bf*)(row + kc + half * 8);
    v8bf hi = *(const v8bf*)(row + kc + 16 + half * 8);
    return __builtin_shufflevector(lo, hi, 0,1,2,3,4,5,6,7,8,9,10,11,12,13,14,15);
}

__device__ __forceinline__ v16bf load_b_frag(const bf16* __restrict__ row, int kc, int half)
{
    v8bf lo = *(const v8bf*)(row + kc + half * 8);
    v8bf hi = *(const v8bf*)(row + kc + 16 + half * 8);
    return __builtin_shufflevector(lo, hi, 0,1,2,3,4,5,6,7,8,9,10,11,12,13,14,15);
}

__device__ __forceinline__ float sigf(float x) { return 1.0f / (1.0f + __expf(-x)); }

// ---------------------------------------------------------------------------
// One LSTM timestep.  grid = 64 blocks (16 hidden cols each), 512 thr (16 waves).
// Wave w: gate g = w>>2, m-tile mt = w&3.  K = 512 (x) + 1024 (h_in).
// ---------------------------------------------------------------------------
__global__ __launch_bounds__(512)
void lstm_step_kernel(const bf16* __restrict__ xemb,   // [T*64, 512]
                      const bf16* __restrict__ wih,    // [4096, 512]
                      const bf16* __restrict__ whh,    // [4096, 1024]
                      const float* __restrict__ bsum,  // [4096]
                      const bf16* __restrict__ h_in,   // [64, 1024] bf16
                      bf16* __restrict__ h_out,        // [64, 1024] bf16
                      float* __restrict__ cst,         // [64, 1024] f32 (in place)
                      float* __restrict__ h_f32,       // [64, 1024] f32
                      int t)
{
    __shared__ float gsm[4][BATCH][16];   // gates staged for elementwise fusion

    const int tid  = threadIdx.x;
    const int lane = tid & 31;
    const int wave = tid >> 5;
    const int g    = wave >> 2;           // gate: 0=i 1=f 2=g 3=o
    const int mt   = wave & 3;            // batch tile (rows mt*16 .. +15)
    const int jb   = blockIdx.x;          // hidden-column block
    const int half = lane >> 4;
    const int l16  = lane & 15;

    const int wrow = g * HIDD + jb * 16 + l16;   // weight row (= output col)
    const int brow = mt * 16 + l16;              // batch row for A operand

    const bf16* __restrict__ xrow   = xemb + ((size_t)t * BATCH + brow) * EMBD;
    const bf16* __restrict__ hrow   = h_in + (size_t)brow * HIDD;
    const bf16* __restrict__ wihrow = wih  + (size_t)wrow * EMBD;
    const bf16* __restrict__ whhrow = whh  + (size_t)wrow * HIDD;

    v8f acc = {};

    // ---- input projection: K = 512 ----
    #pragma unroll 4
    for (int kc = 0; kc < EMBD; kc += 32) {
        v16bf a = load_a_frag(xrow,   kc, half);
        v16bf b = load_b_frag(wihrow, kc, half);
        acc = __builtin_amdgcn_wmma_f32_16x16x32_bf16(false, a, false, b,
                                                      (short)0, acc, false, false);
    }
    // ---- recurrent projection: K = 1024 ----
    #pragma unroll 4
    for (int kc = 0; kc < HIDD; kc += 32) {
        v16bf a = load_a_frag(hrow,   kc, half);
        v16bf b = load_b_frag(whhrow, kc, half);
        acc = __builtin_amdgcn_wmma_f32_16x16x32_bf16(false, a, false, b,
                                                      (short)0, acc, false, false);
    }

    // fused bias (b_ih + b_hh), depends only on output column = l16's wrow
    const float bb = bsum[wrow];
    #pragma unroll
    for (int r = 0; r < 8; ++r) acc[r] += bb;

    // D layout: lane L, VGPR r -> M = half*8 + r, N = l16
    #pragma unroll
    for (int r = 0; r < 8; ++r)
        gsm[g][mt * 16 + half * 8 + r][l16] = acc[r];

    __syncthreads();

    // ---- elementwise LSTM cell update: 64x16 elems over 512 threads ----
    #pragma unroll
    for (int e = tid; e < BATCH * 16; e += 512) {
        const int b_ = e >> 4;
        const int n  = e & 15;
        float iv = sigf(gsm[0][b_][n]);
        float fv = sigf(gsm[1][b_][n]);
        float gv = tanhf(gsm[2][b_][n]);
        float ov = sigf(gsm[3][b_][n]);
        const int idx = b_ * HIDD + jb * 16 + n;
        float cn = fv * cst[idx] + iv * gv;
        float hn = ov * tanhf(cn);
        cst[idx]   = cn;
        h_f32[idx] = hn;
        h_out[idx] = (bf16)hn;
    }
}

// ---------------------------------------------------------------------------
// Final packing: out = [h (output), h, c] each [1, 64, 1024] f32.
// ---------------------------------------------------------------------------
__global__ void finalize_kernel(const float* __restrict__ h,
                                const float* __restrict__ c,
                                float* __restrict__ out)
{
    int i = blockIdx.x * blockDim.x + threadIdx.x;
    if (i < BATCH * HIDD) {
        float hv = h[i];
        out[i]                    = hv;
        out[BATCH * HIDD + i]     = hv;
        out[2 * BATCH * HIDD + i] = c[i];
    }
}

// ---------------------------------------------------------------------------
extern "C" void kernel_launch(void* const* d_in, const int* in_sizes, int n_in,
                              void* d_out, int out_size, void* d_ws, size_t ws_size,
                              hipStream_t stream)
{
    const int*   sent   = (const int*)d_in[0];
    const float* table  = (const float*)d_in[1];
    const float* wih_f  = (const float*)d_in[2];
    const float* whh_f  = (const float*)d_in[3];
    const float* bih    = (const float*)d_in[4];
    const float* bhh    = (const float*)d_in[5];
    float*       out    = (float*)d_out;

    // workspace carve (256B aligned)
    char* p = (char*)d_ws;
    auto carve = [&](size_t bytes) -> char* {
        char* r = p;
        p += (bytes + 255) & ~(size_t)255;
        return r;
    };
    bf16*  wih_b = (bf16*)carve((size_t)G4 * EMBD * sizeof(bf16));        //  4 MB
    bf16*  whh_b = (bf16*)carve((size_t)G4 * HIDD * sizeof(bf16));        //  8 MB
    float* bsum  = (float*)carve((size_t)G4 * sizeof(float));             // 16 KB
    bf16*  xemb  = (bf16*)carve((size_t)TLEN * BATCH * EMBD * sizeof(bf16)); // 32 MB
    bf16*  hb0   = (bf16*)carve((size_t)BATCH * HIDD * sizeof(bf16));
    bf16*  hb1   = (bf16*)carve((size_t)BATCH * HIDD * sizeof(bf16));
    float* cst   = (float*)carve((size_t)BATCH * HIDD * sizeof(float));
    float* hf32  = (float*)carve((size_t)BATCH * HIDD * sizeof(float));

    prep_kernel<<<512, 256, 0, stream>>>(wih_f, whh_f, bih, bhh,
                                         wih_b, whh_b, bsum, hb0, cst);

    embed_kernel<<<TLEN * BATCH, 128, 0, stream>>>(sent, table, xemb);

    bf16* hp[2] = { hb0, hb1 };
    for (int t = 0; t < TLEN; ++t) {
        lstm_step_kernel<<<HIDD / 16, 512, 0, stream>>>(
            xemb, wih_b, whh_b, bsum,
            hp[t & 1], hp[(t + 1) & 1], cst, hf32, t);
    }

    finalize_kernel<<<(BATCH * HIDD + 255) / 256, 256, 0, stream>>>(hf32, cst, out);
}
